// ContrastiveLoss_70849780515159
// MI455X (gfx1250) — matrix operand order repaired and checked by reading, not compiled, not run on hardware
//
#include <hip/hip_runtime.h>

// ContrastiveLoss fused GEMM (sim = X X^T) + masked hinge reduction, MI455X / gfx1250.
//
// Roofline: 137 GFLOP GEMM, 32 MB inputs (L2-resident; L2 = 192 MB) -> compute bound.
// Precision: split-bf16 (x = hi + lo), sim ~= hi*hi + hi*lo + lo*hi via
//            v_wmma_f32_16x16x32_bf16 -- near-fp32 accuracy at bf16 matrix rates.
// Symmetry:  only lower-triangle tiles; off-diagonal elements weighted 2x.
// Tiling:    32x32 output per wave = 2x2 WMMA subtiles -> 4 independent
//            accumulator chains; 12 WMMAs per 16 b128 loads per K=32 step.
// Unroll:    k-loop fully unrolled so every load's k-offset folds into the
//            24-bit instruction immediate -- zero steady-state address VALU,
//            no WAR-hazard v_nops into WMMA operand registers, and the
//            scheduler is free to hoist loads as far ahead as registers allow.
//
// Workspace: [Xhi: 16 MB bf16][Xlo: 16 MB bf16][partials: GRID_MAIN floats]

#define N_ROWS 8192
#define DIM 1024
#define MARGIN_F 0.3f
#define MACRO 32                                        // per-wave output tile edge
#define MT_1D (N_ROWS / MACRO)                          // 256
#define NUM_MACRO_TILES (MT_1D * (MT_1D + 1) / 2)       // 32896
#define WAVES_PER_BLOCK 8
#define GRID_MAIN (NUM_MACRO_TILES / WAVES_PER_BLOCK)   // 4112 (exact)

typedef __attribute__((ext_vector_type(4)))  __bf16 bf16x4;
typedef __attribute__((ext_vector_type(8)))  __bf16 bf16x8;
typedef __attribute__((ext_vector_type(16))) __bf16 bf16x16;
typedef __attribute__((ext_vector_type(4)))  float  f32x4;
typedef __attribute__((ext_vector_type(8)))  float  f32x8;

// ---------------- Phase 0: split fp32 into bf16 hi + bf16 lo ----------------
__global__ void __launch_bounds__(256)
cl_split_bf16_kernel(const float* __restrict__ x,
                     __bf16* __restrict__ hi,
                     __bf16* __restrict__ lo) {
    const int idx = (blockIdx.x * blockDim.x + threadIdx.x) * 4;
    f32x4 v = *(const f32x4*)(x + idx);
    bf16x4 h, l;
#pragma unroll
    for (int e = 0; e < 4; ++e) {
        float f = v[e];
        __bf16 hb = (__bf16)f;                 // round-to-nearest-even
        float resid = f - (float)hb;
        h[e] = hb;
        l[e] = (__bf16)resid;
    }
    *(bf16x4*)(hi + idx) = h;
    *(bf16x4*)(lo + idx) = l;
}

// ---------------- Phase 1: fused WMMA GEMM + masked-loss epilogue ----------------
__global__ void __launch_bounds__(256)
cl_contrastive_wmma_kernel(const __bf16* __restrict__ Xhi,
                           const __bf16* __restrict__ Xlo,
                           const int* __restrict__ targets,
                           float* __restrict__ partials) {
    __shared__ float wave_sums[WAVES_PER_BLOCK];

    const int wave = threadIdx.x >> 5;
    const int lane = threadIdx.x & 31;
    const int t = blockIdx.x * WAVES_PER_BLOCK + wave;

    // Decode linear lower-triangle macro-tile index t -> (a, b), b <= a.
    int a = (int)((sqrtf(8.0f * (float)t + 1.0f) - 1.0f) * 0.5f);
    while ((a + 1) * (a + 2) / 2 <= t) ++a;   // wave-uniform fixup
    while (a * (a + 1) / 2 > t) --a;
    const int b = t - a * (a + 1) / 2;
    const int ibase = b * MACRO;   // M macro-tile base (rows i)
    const int jbase = a * MACRO;   // N macro-tile base (rows j), jbase >= ibase

    // WMMA bf16 operand k order chosen here (wave32; contiguous, not the ISA table order; A and B match):
    //  A (16x32, MxK): lanes 0-15 row M=lane K=0..15;
    //                  lanes 16-31 row M=lane-16, K=16..31.
    //  B (32x16, KxN): lanes 0-15 col N=lane K=0..15; lanes 16-31 col N=lane-16 K=16..31.
    const int half = lane >> 4;
    const int lrow = lane & 15;

    const __bf16* Ahi[2];
    const __bf16* Alo[2];
    const __bf16* Bhi[2];
    const __bf16* Blo[2];
#pragma unroll
    for (int s = 0; s < 2; ++s) {
        Ahi[s] = Xhi + (size_t)(ibase + s * 16 + lrow) * DIM + half * 16;
        Alo[s] = Xlo + (size_t)(ibase + s * 16 + lrow) * DIM + half * 16;
        Bhi[s] = Xhi + (size_t)(jbase + s * 16 + lrow) * DIM + half * 16;
        Blo[s] = Xlo + (size_t)(jbase + s * 16 + lrow) * DIM + half * 16;
    }

    f32x8 acc[2][2];
#pragma unroll
    for (int si = 0; si < 2; ++si)
#pragma unroll
        for (int sj = 0; sj < 2; ++sj)
            acc[si][sj] = (f32x8){0.f, 0.f, 0.f, 0.f, 0.f, 0.f, 0.f, 0.f};

    union bfu { bf16x16 v; bf16x8 h[2]; };

    // Fully unrolled k-loop: constant k -> all load offsets become instruction
    // immediates off loop-invariant base address VGPRs.
#pragma unroll
    for (int k = 0; k < DIM; k += 32) {
        bfu ah[2], al[2];
        bf16x16 bh[2], bl[2];
#pragma unroll
        for (int s = 0; s < 2; ++s) {
            ah[s].h[0] = *(const bf16x8*)(Ahi[s] + k);
            ah[s].h[1] = *(const bf16x8*)(Ahi[s] + k + 8);
            al[s].h[0] = *(const bf16x8*)(Alo[s] + k);
            al[s].h[1] = *(const bf16x8*)(Alo[s] + k + 8);
            bh[s] = *(const bf16x16*)(Bhi[s] + k);
            bl[s] = *(const bf16x16*)(Blo[s] + k);
        }

        // Three passes interleaved across the 4 independent accumulator chains,
        // so same-chain dependent WMMAs are >= 4 instructions apart.
#pragma unroll
        for (int si = 0; si < 2; ++si)
#pragma unroll
            for (int sj = 0; sj < 2; ++sj)
                acc[si][sj] = __builtin_amdgcn_wmma_f32_16x16x32_bf16(
                    false, ah[si].v, false, bh[sj], (short)0, acc[si][sj], false, false);
#pragma unroll
        for (int si = 0; si < 2; ++si)
#pragma unroll
            for (int sj = 0; sj < 2; ++sj)
                acc[si][sj] = __builtin_amdgcn_wmma_f32_16x16x32_bf16(
                    false, ah[si].v, false, bl[sj], (short)0, acc[si][sj], false, false);
#pragma unroll
        for (int si = 0; si < 2; ++si)
#pragma unroll
            for (int sj = 0; sj < 2; ++sj)
                acc[si][sj] = __builtin_amdgcn_wmma_f32_16x16x32_bf16(
                    false, al[si].v, false, bh[sj], (short)0, acc[si][sj], false, false);
    }

    // Epilogue. C/D layout: VGPR r, lanes 0-15 -> (M=r, N=lane); lanes 16-31 -> (M=r+8, N=lane-16).
    const int moff = half * 8;
    int tj[2], gj[2];
#pragma unroll
    for (int sj = 0; sj < 2; ++sj) {
        gj[sj] = jbase + sj * 16 + lrow;
        tj[sj] = targets[gj[sj]];
    }

    float local = 0.f;
#pragma unroll
    for (int si = 0; si < 2; ++si) {
#pragma unroll
        for (int r = 0; r < 8; ++r) {
            const int gi = ibase + si * 16 + moff + r;
            const int ti = targets[gi];
#pragma unroll
            for (int sj = 0; sj < 2; ++sj) {
                const float s = acc[si][sj][r];
                const float f = (ti == tj[sj]) ? ((s < 1.0f) ? (1.0f - s) : 0.0f)
                                               : ((s > MARGIN_F) ? s : 0.0f);
                // symmetry weighting at element granularity
                const float w = (gj[sj] > gi) ? 2.0f : ((gj[sj] == gi) ? 1.0f : 0.0f);
                local = fmaf(w, f, local);
            }
        }
    }

    // Deterministic wave32 reduction.
#pragma unroll
    for (int off = 16; off > 0; off >>= 1)
        local += __shfl_xor(local, off, 32);

    if (lane == 0) wave_sums[wave] = local;
    __syncthreads();
    if (threadIdx.x == 0) {
        float s = 0.f;
#pragma unroll
        for (int wv = 0; wv < WAVES_PER_BLOCK; ++wv) s += wave_sums[wv];
        partials[blockIdx.x] = s;
    }
}

// ---------------- Phase 2: deterministic final reduction ----------------
__global__ void __launch_bounds__(256)
cl_reduce_kernel(const float* __restrict__ partials, int n, float* __restrict__ out) {
    __shared__ float sm[256];
    float s = 0.f;
    for (int i = threadIdx.x; i < n; i += 256) s += partials[i];  // fixed order
    sm[threadIdx.x] = s;
    __syncthreads();
    for (int stride = 128; stride > 0; stride >>= 1) {
        if ((int)threadIdx.x < stride) sm[threadIdx.x] += sm[threadIdx.x + stride];
        __syncthreads();
    }
    if (threadIdx.x == 0) out[0] = sm[0] / (float)N_ROWS;
}

extern "C" void kernel_launch(void* const* d_in, const int* in_sizes, int n_in,
                              void* d_out, int out_size, void* d_ws, size_t ws_size,
                              hipStream_t stream) {
    const float* X       = (const float*)d_in[0];
    const int*   targets = (const int*)d_in[1];
    float*       out     = (float*)d_out;

    __bf16* Xhi = (__bf16*)d_ws;
    __bf16* Xlo = Xhi + (size_t)N_ROWS * DIM;
    float*  partials = (float*)(Xlo + (size_t)N_ROWS * DIM);

    // Phase 0: fp32 -> bf16 hi/lo split (4 elements per thread).
    const int total = N_ROWS * DIM;
    cl_split_bf16_kernel<<<total / (256 * 4), 256, 0, stream>>>(X, Xhi, Xlo);

    // Phase 1: fused symmetric GEMM + loss, one 32x32 macro-tile per wave.
    cl_contrastive_wmma_kernel<<<GRID_MAIN, 256, 0, stream>>>(Xhi, Xlo, targets, partials);

    // Phase 2: scalar output.
    cl_reduce_kernel<<<1, 256, 0, stream>>>(partials, GRID_MAIN, out);
}
